// FP8StaticLinearQuantizer_49400713839125
// MI455X (gfx1250) — compile-verified, matrix operand order checked by reading, not run on hardware
//
#include <hip/hip_runtime.h>
#include <hip/hip_bf16.h>
#include <math.h>

// Problem dims (from reference setup_inputs): B=4, S=2048, Din=Dout=4096
#define M_TOTAL 8192
#define D_IN    4096
#define D_OUT   4096

typedef int   v4i  __attribute__((ext_vector_type(4)));
typedef int   v16i __attribute__((ext_vector_type(16)));
typedef float v8f  __attribute__((ext_vector_type(8)));

// pointer-to-AS1(global) v4i and pointer-to-AS3(LDS) v4i, as the async builtin expects
typedef __attribute__((address_space(1))) v4i* glob_v4i_ptr;
typedef __attribute__((address_space(3))) v4i* lds_v4i_ptr;

#if __has_builtin(__builtin_amdgcn_global_load_async_to_lds_b128)
#define HAVE_ASYNC_LDS 1
#else
#define HAVE_ASYNC_LDS 0
#endif

__device__ __forceinline__ void wait_asynccnt0() {
#if __has_builtin(__builtin_amdgcn_s_wait_asynccnt)
    __builtin_amdgcn_s_wait_asynccnt(0);
#else
    asm volatile("s_wait_asynccnt 0x0" ::: "memory");
#endif
}

// ---------------- fp8 e4m3fn helpers (manual, RNE, saturating at 448) -------
__device__ __forceinline__ unsigned f32_to_e4m3(float x) {
    unsigned ubits = __float_as_uint(x);
    unsigned sign  = (ubits >> 24) & 0x80u;
    float ax = fabsf(x);
    if (!(ax > 0.0f))  return sign;            // zero (NaN -> 0, inputs are clipped)
    if (ax >= 448.0f)  return sign | 0x7Eu;    // saturate to max normal = 448
    unsigned u = __float_as_uint(ax);
    int e = (int)(u >> 23) - 127;
    if (e < -6) {                              // subnormal range, step = 2^-9
        int m = (int)rintf(ax * 512.0f);
        if (m >= 8) return sign | 0x08u;       // rounds up to min normal
        return sign | (unsigned)m;
    }
    unsigned mant = u & 0x7FFFFFu;
    unsigned lsb  = (mant >> 20) & 1u;
    mant += 0x7FFFFu + lsb;                    // round-to-nearest-even to 3 bits
    if (mant >> 23) { mant = 0; ++e; }
    if (e > 8) return sign | 0x7Eu;
    return sign | ((unsigned)(e + 7) << 3) | (mant >> 20);
}

__device__ __forceinline__ float e4m3_to_f32(unsigned b) {
    unsigned exp = (b >> 3) & 0xFu;
    unsigned man = b & 7u;
    float v;
    if (exp == 0) v = (float)man * (1.0f / 512.0f);               // subnormal
    else          v = __uint_as_float(((exp + 120u) << 23) | (man << 20));
    return (b & 0x80u) ? -v : v;
}

__device__ __forceinline__ float scale_from_bits(unsigned amax_bits) {
    float amax = __uint_as_float(amax_bits);   // abs-max stored as raw f32 bits
    return fmaxf(amax * (1.0f / 448.0f), 1e-12f);
}

// ---------------- kernel 1: init atomic slots --------------------------------
__global__ void init_kernel(unsigned* amax_in, unsigned* amax_out) {
    *amax_in = 0u;
    *amax_out = 0u;
}

// ---------------- kernel 2: abs-max over x ----------------------------------
__global__ __launch_bounds__(256) void absmax_kernel(const float4* __restrict__ x4,
                                                     int n4, unsigned* __restrict__ amax) {
    __shared__ float sm[256];
    float m = 0.0f;
    for (int i = blockIdx.x * 256 + threadIdx.x; i < n4; i += gridDim.x * 256) {
        float4 v = x4[i];
        m = fmaxf(m, fmaxf(fmaxf(fabsf(v.x), fabsf(v.y)),
                           fmaxf(fabsf(v.z), fabsf(v.w))));
    }
    sm[threadIdx.x] = m;
    __syncthreads();
    for (int s = 128; s > 0; s >>= 1) {
        if (threadIdx.x < (unsigned)s) sm[threadIdx.x] = fmaxf(sm[threadIdx.x], sm[threadIdx.x + s]);
        __syncthreads();
    }
    if (threadIdx.x == 0) atomicMax(amax, __float_as_uint(sm[0]));
}

// ---------------- kernel 3a: quantize activations ---------------------------
__global__ __launch_bounds__(256) void quantize_x_kernel(const float4* __restrict__ x4,
                                                         const unsigned* __restrict__ amax_in,
                                                         unsigned* __restrict__ q4, int n4) {
    int i = blockIdx.x * 256 + threadIdx.x;
    if (i >= n4) return;
    float inv = 1.0f / scale_from_bits(*amax_in);
    float4 v = x4[i];
    unsigned b0 = f32_to_e4m3(v.x * inv);
    unsigned b1 = f32_to_e4m3(v.y * inv);
    unsigned b2 = f32_to_e4m3(v.z * inv);
    unsigned b3 = f32_to_e4m3(v.w * inv);
    q4[i] = b0 | (b1 << 8) | (b2 << 16) | (b3 << 24);
}

// ---------------- kernel 3b: pack weight (already fp8-grid f32) to bytes ----
__global__ __launch_bounds__(256) void quantize_w_kernel(const float4* __restrict__ w4,
                                                         unsigned* __restrict__ q4, int n4) {
    int i = blockIdx.x * 256 + threadIdx.x;
    if (i >= n4) return;
    float4 v = w4[i];
    unsigned b0 = f32_to_e4m3(v.x);
    unsigned b1 = f32_to_e4m3(v.y);
    unsigned b2 = f32_to_e4m3(v.z);
    unsigned b3 = f32_to_e4m3(v.w);
    q4[i] = b0 | (b1 << 8) | (b2 << 16) | (b3 << 24);
}

// ---------------- kernel 4: fp8 WMMA GEMM -----------------------------------
// C[M,N] = (qa[M,K] . qw[N,K]^T) * (in_scale*w_scale) + bias
// Block tile 256x128, K-step 128. 256 threads = 8 waves; wave w owns rows
// [w*32, w*32+32) as two 16-row subtiles x 128 cols => 16 accumulators.
// Each B fragment feeds 2 WMMAs (halves LDS fragment bandwidth per FLOP).
// Double-buffered LDS, async global->LDS staging overlapped with WMMA.
#define BM 256
#define BN 128
#define BK 128
#define LDSS 144   // LDS row stride (16B aligned, conflict-free fragment reads)

union Frag { int4 q[4]; v16i v; };

__global__ __launch_bounds__(256) void gemm_fp8_kernel(
    const unsigned char* __restrict__ qa,
    const unsigned char* __restrict__ qw,
    const float* __restrict__ bias,
    const unsigned* __restrict__ amax_in,
    const float* __restrict__ wscale_p,
    float* __restrict__ out,
    unsigned* __restrict__ amax_out)
{
    __shared__ unsigned char ldsA[2][BM * LDSS];   // 2 x 36 KB
    __shared__ unsigned char ldsB[2][BN * LDSS];   // 2 x 18 KB
    __shared__ float redmax[256];

    const int t    = threadIdx.x;
    const int lane = t & 31;
    const int wv   = t >> 5;                 // wave id 0..7
    const int m0   = blockIdx.y * BM;
    const int n0   = blockIdx.x * BN;

    const int lr = lane & 15;                // fragment row within 16
    const int lk = (lane >> 4) * 64;         // K byte offset: halves of 128

    // staging: A -> each thread copies a full 128B row (rows 0..255)
    //          B -> 2 threads per row, 64B each (rows 0..127)
    const unsigned char* gArow = qa + (size_t)(m0 + t) * D_IN;
    const unsigned char* gBrow = qw + (size_t)(n0 + (t >> 1)) * D_IN + (t & 1) * 64;
    const int ldsA_off = t * LDSS;
    const int ldsB_off = (t >> 1) * LDSS + (t & 1) * 64;

    auto stage = [&](int buf, int kt) {
#if HAVE_ASYNC_LDS
        #pragma unroll
        for (int j = 0; j < 8; ++j) {
            __builtin_amdgcn_global_load_async_to_lds_b128(
                (glob_v4i_ptr)(gArow + kt + j * 16),
                (lds_v4i_ptr)&ldsA[buf][ldsA_off + j * 16], 0, 0);
        }
        #pragma unroll
        for (int j = 0; j < 4; ++j) {
            __builtin_amdgcn_global_load_async_to_lds_b128(
                (glob_v4i_ptr)(gBrow + kt + j * 16),
                (lds_v4i_ptr)&ldsB[buf][ldsB_off + j * 16], 0, 0);
        }
#else
        const int4* gA = reinterpret_cast<const int4*>(gArow + kt);
        int4* sA = reinterpret_cast<int4*>(&ldsA[buf][ldsA_off]);
        #pragma unroll
        for (int j = 0; j < 8; ++j) sA[j] = gA[j];
        const int4* gB = reinterpret_cast<const int4*>(gBrow + kt);
        int4* sB = reinterpret_cast<int4*>(&ldsB[buf][ldsB_off]);
        #pragma unroll
        for (int j = 0; j < 4; ++j) sB[j] = gB[j];
#endif
    };

    v8f acc0[8], acc1[8];
    #pragma unroll
    for (int i = 0; i < 8; ++i) {
        v8f z = {0.f, 0.f, 0.f, 0.f, 0.f, 0.f, 0.f, 0.f};
        acc0[i] = z; acc1[i] = z;
    }

    // prologue: stage tile 0
    stage(0, 0);
    wait_asynccnt0();
    __syncthreads();

    int buf = 0;
    for (int kt = 0; kt < D_IN; kt += BK, buf ^= 1) {
        // kick off async staging of the next tile into the other buffer
        if (kt + BK < D_IN) stage(buf ^ 1, kt + BK);

        // two A fragments: rows [wv*32, +16) and [wv*32+16, +16)
        Frag af0, af1;
        {
            const int4* p0 = reinterpret_cast<const int4*>(
                &ldsA[buf][(wv * 32 + lr) * LDSS + lk]);
            af0.q[0] = p0[0]; af0.q[1] = p0[1]; af0.q[2] = p0[2]; af0.q[3] = p0[3];
            const int4* p1 = reinterpret_cast<const int4*>(
                &ldsA[buf][(wv * 32 + 16 + lr) * LDSS + lk]);
            af1.q[0] = p1[0]; af1.q[1] = p1[1]; af1.q[2] = p1[2]; af1.q[3] = p1[3];
        }

        // 2-deep pipelined B fragments over the 8 N-tiles; each feeds 2 WMMAs
        Frag bf[2];
        {
            const int4* pB = reinterpret_cast<const int4*>(&ldsB[buf][lr * LDSS + lk]);
            bf[0].q[0] = pB[0]; bf[0].q[1] = pB[1]; bf[0].q[2] = pB[2]; bf[0].q[3] = pB[3];
        }
        #pragma unroll
        for (int nt = 0; nt < 8; ++nt) {
            if (nt < 7) {
                const int4* pB = reinterpret_cast<const int4*>(
                    &ldsB[buf][((nt + 1) * 16 + lr) * LDSS + lk]);
                Frag& nx = bf[(nt + 1) & 1];
                nx.q[0] = pB[0]; nx.q[1] = pB[1]; nx.q[2] = pB[2]; nx.q[3] = pB[3];
            }
            acc0[nt] = __builtin_amdgcn_wmma_f32_16x16x128_fp8_fp8(
                af0.v, bf[nt & 1].v, (short)0, acc0[nt], false, false);
            acc1[nt] = __builtin_amdgcn_wmma_f32_16x16x128_fp8_fp8(
                af1.v, bf[nt & 1].v, (short)0, acc1[nt], false, false);
        }

        // next-tile async writes must land before anyone reads that buffer
        wait_asynccnt0();
        __syncthreads();
    }

    // epilogue: dequant-scale, bias, store, output abs-max
    const float s = scale_from_bits(*amax_in) * wscale_p[0];
    float lmax = 0.0f;
    const int mrow0 = m0 + wv * 32 + ((lane >> 4) << 3);  // C layout: M = r + 8*(lane/16)
    #pragma unroll
    for (int nt = 0; nt < 8; ++nt) {
        const int col = n0 + nt * 16 + lr;
        const float bv = bias[col];
        #pragma unroll
        for (int r = 0; r < 8; ++r) {
            float v0 = acc0[nt][r] * s + bv;
            float v1 = acc1[nt][r] * s + bv;
            lmax = fmaxf(lmax, fmaxf(fabsf(v0), fabsf(v1)));
            out[(size_t)(mrow0 + r) * D_OUT + col] = v0;
            out[(size_t)(mrow0 + 16 + r) * D_OUT + col] = v1;
        }
    }
    redmax[t] = lmax;
    __syncthreads();
    for (int sred = 128; sred > 0; sred >>= 1) {
        if (t < sred) redmax[t] = fmaxf(redmax[t], redmax[t + sred]);
        __syncthreads();
    }
    if (t == 0) atomicMax(amax_out, __float_as_uint(redmax[0]));
}

// ---------------- kernel 5: requantize output in place ----------------------
__global__ __launch_bounds__(256) void requant_out_kernel(float4* __restrict__ o4,
                                                          const unsigned* __restrict__ amax_out,
                                                          int n4) {
    int i = blockIdx.x * 256 + threadIdx.x;
    if (i >= n4) return;
    float sc  = scale_from_bits(*amax_out);
    float inv = 1.0f / sc;
    float4 v = o4[i];
    v.x = e4m3_to_f32(f32_to_e4m3(v.x * inv)) * sc;
    v.y = e4m3_to_f32(f32_to_e4m3(v.y * inv)) * sc;
    v.z = e4m3_to_f32(f32_to_e4m3(v.z * inv)) * sc;
    v.w = e4m3_to_f32(f32_to_e4m3(v.w * inv)) * sc;
    o4[i] = v;
}

// ---------------- host-side launcher ----------------------------------------
extern "C" void kernel_launch(void* const* d_in, const int* in_sizes, int n_in,
                              void* d_out, int out_size, void* d_ws, size_t ws_size,
                              hipStream_t stream) {
    const float* x     = (const float*)d_in[0];   // [4,2048,4096] f32
    const float* qwf   = (const float*)d_in[1];   // [4096,4096] f32 (fp8-grid values)
    const float* wsc   = (const float*)d_in[2];   // scalar
    const float* bias  = (const float*)d_in[3];   // [4096]
    float* out = (float*)d_out;                   // [4,2048,4096] f32

    unsigned char* ws = (unsigned char*)d_ws;
    unsigned* amax_in  = (unsigned*)ws;
    unsigned* amax_out = (unsigned*)(ws + 4);
    unsigned char* qa  = ws + 256;                                  // 32 MB fp8 activations
    unsigned char* qwb = ws + 256 + (size_t)M_TOTAL * D_IN;         // 16 MB fp8 weights

    const int NX = M_TOTAL * D_IN;   // 33,554,432
    const int NW = D_OUT * D_IN;     // 16,777,216

    init_kernel<<<1, 1, 0, stream>>>(amax_in, amax_out);
    absmax_kernel<<<1024, 256, 0, stream>>>((const float4*)x, NX / 4, amax_in);
    quantize_x_kernel<<<(NX / 4 + 255) / 256, 256, 0, stream>>>(
        (const float4*)x, amax_in, (unsigned*)qa, NX / 4);
    quantize_w_kernel<<<(NW / 4 + 255) / 256, 256, 0, stream>>>(
        (const float4*)qwf, (unsigned*)qwb, NW / 4);

    dim3 grid(D_OUT / BN, M_TOTAL / BM);   // 32 x 32 = 1024 blocks
    gemm_fp8_kernel<<<grid, 256, 0, stream>>>(qa, qwb, bias, amax_in, wsc, out, amax_out);

    requant_out_kernel<<<(NX / 4 + 255) / 256, 256, 0, stream>>>(
        (float4*)out, amax_out, NX / 4);
}
